// Attention_SelfMask_5102421148129
// MI455X (gfx1250) — compile-verified, matrix operand order checked by reading, not run on hardware
//
#include <hip/hip_runtime.h>

#define BATCH 8
#define SEQ 1024
#define DIM 1024
#define NHEAD 16
#define HDIM 64
#define ATT_SCALE 0.125f  /* 64^-0.5 */

typedef __bf16 bf16_t;
typedef __attribute__((ext_vector_type(16))) __bf16 v16bf;
typedef __attribute__((ext_vector_type(8)))  __bf16 v8bf;
typedef __attribute__((ext_vector_type(8)))  float  v8f;

// D = A(16x32) * B(32x16)^layout + C   (bf16 in, f32 accum)
__device__ __forceinline__ v8f wmma_bf16(v16bf a, v16bf b, v8f c) {
  return __builtin_amdgcn_wmma_f32_16x16x32_bf16(false, a, false, b, (short)0, c,
                                                 false, false);
}

// Per-lane fragment load for the CDNA5 16-bit A/B layout:
// vector elems 0..7  <- row[kLo .. kLo+7]   (one b128)
// vector elems 8..15 <- row[kHi .. kHi+7]   (one b128)
// where kLo = k0 + half*8, kHi = k0 + 16 + half*8 (half = lane>>4).
__device__ __forceinline__ v16bf load_frag(const bf16_t* rowptr, int kLo, int kHi) {
  v8bf lo = *reinterpret_cast<const v8bf*>(rowptr + kLo);
  v8bf hi = *reinterpret_cast<const v8bf*>(rowptr + kHi);
  v16bf r;
#pragma unroll
  for (int i = 0; i < 8; ++i) { r[i] = lo[i]; r[i + 8] = hi[i]; }
  return r;
}

__device__ __forceinline__ v8f zero8() {
  v8f z;
#pragma unroll
  for (int i = 0; i < 8; ++i) z[i] = 0.0f;
  return z;
}

// ---------------- fp32 -> bf16 convert ----------------
__global__ void cvt_f32_bf16(const float* __restrict__ in, bf16_t* __restrict__ out, int n) {
  int i = blockIdx.x * blockDim.x + threadIdx.x;
  if (i < n) out[i] = (bf16_t)in[i];
}

// ---------------- QKV GEMM: [8192,1024] x [3072,1024]^T, scatter epilogue ----
// 4 waves/WG, each wave: 32 rows x 64 cols. grid = (64, 48).
__global__ void __launch_bounds__(128) qkv_gemm(
    const bf16_t* __restrict__ xb,    // [8192,1024]
    const bf16_t* __restrict__ wq,    // [3072,1024]
    bf16_t* __restrict__ qout,        // [B,H,N,D]  (pre-scaled)
    bf16_t* __restrict__ kout,        // [B,H,N,D]
    bf16_t* __restrict__ vtout) {     // [B,H,D,N]
  const int lane = threadIdx.x & 31;
  const int wave = threadIdx.x >> 5;
  const int half = lane >> 4;
  const int l16  = lane & 15;
  const int m0 = blockIdx.x * 128 + wave * 32;
  const int n0 = blockIdx.y * 64;

  v8f acc[2][4];
#pragma unroll
  for (int i = 0; i < 2; ++i)
#pragma unroll
    for (int j = 0; j < 4; ++j) acc[i][j] = zero8();

  for (int k0 = 0; k0 < DIM; k0 += 32) {
    const int kLo = k0 + half * 8, kHi = k0 + 16 + half * 8;
    v16bf a0 = load_frag(xb + (size_t)(m0 + l16) * DIM,      kLo, kHi);
    v16bf a1 = load_frag(xb + (size_t)(m0 + 16 + l16) * DIM, kLo, kHi);
#pragma unroll
    for (int ct = 0; ct < 4; ++ct) {
      v16bf bf = load_frag(wq + (size_t)(n0 + ct * 16 + l16) * DIM, kLo, kHi);
      acc[0][ct] = wmma_bf16(a0, bf, acc[0][ct]);
      acc[1][ct] = wmma_bf16(a1, bf, acc[1][ct]);
    }
  }

  // n0 is 64-aligned -> this block is exactly one (s, h) pair.
  const int s = n0 >> 10;             // 0=q, 1=k, 2=v
  const int h = (n0 & 1023) >> 6;
#pragma unroll
  for (int rt = 0; rt < 2; ++rt) {
#pragma unroll
    for (int ct = 0; ct < 4; ++ct) {
#pragma unroll
      for (int r = 0; r < 8; ++r) {
        const int m   = m0 + rt * 16 + r + half * 8;
        const int b   = m >> 10;
        const int tok = m & 1023;
        const int d   = ct * 16 + l16;
        const float v = acc[rt][ct][r];
        if (s == 0)
          qout[((size_t)(b * NHEAD + h) * SEQ + tok) * HDIM + d] = (bf16_t)(v * ATT_SCALE);
        else if (s == 1)
          kout[((size_t)(b * NHEAD + h) * SEQ + tok) * HDIM + d] = (bf16_t)v;
        else
          vtout[((size_t)(b * NHEAD + h) * HDIM + d) * SEQ + tok] = (bf16_t)v;
      }
    }
  }
}

// ---------------- Flash attention ----------------
// grid = B*H*(N/128) = 1024 blocks, 256 threads (8 waves).
// Each wave: 16 query rows, loops over key blocks of 32, online softmax.
__global__ void __launch_bounds__(256) flash_attn(
    const bf16_t* __restrict__ q,    // [B,H,N,D], scaled
    const bf16_t* __restrict__ k,    // [B,H,N,D]
    const bf16_t* __restrict__ vt,   // [B,H,D,N]
    const float*  __restrict__ mask, // [B,N,N]
    bf16_t* __restrict__ out) {      // [B,N,C]
  __shared__ __align__(16) bf16_t pLds[8][16 * 32];

  const int lane = threadIdx.x & 31;
  const int wave = threadIdx.x >> 5;
  const int half = lane >> 4;
  const int l16  = lane & 15;
  const int bh   = blockIdx.x >> 3;    // b*H + h
  const int qblk = blockIdx.x & 7;
  const int b    = bh >> 4;
  const int h    = bh & 15;
  const int q0   = qblk * 128 + wave * 16;

  // Q fragments: rows q0+l16, K(=head dim) 0..63 as two K=32 fragments
  const bf16_t* qrow = q + ((size_t)bh * SEQ + q0 + l16) * HDIM;
  v16bf qa0 = load_frag(qrow, half * 8,      16 + half * 8);
  v16bf qa1 = load_frag(qrow, 32 + half * 8, 48 + half * 8);

  const bf16_t* kbase = k  + (size_t)bh * SEQ * HDIM;
  const bf16_t* vbase = vt + (size_t)bh * HDIM * SEQ;
  const float*  mbase = mask + (size_t)b * SEQ * SEQ;

  float mrow[8], lrow[8];
  v8f acc[4];
#pragma unroll
  for (int r = 0; r < 8; ++r) { mrow[r] = -1e30f; lrow[r] = 0.0f; }
#pragma unroll
  for (int dt = 0; dt < 4; ++dt) acc[dt] = zero8();

  for (int kb = 0; kb < SEQ; kb += 32) {
    // S = Q * K^T for two 16-key tiles
    v8f sc[2];
#pragma unroll
    for (int ct = 0; ct < 2; ++ct) {
      const bf16_t* krow = kbase + (size_t)(kb + ct * 16 + l16) * HDIM;
      v16bf kf0 = load_frag(krow, half * 8,      16 + half * 8);
      v16bf kf1 = load_frag(krow, 32 + half * 8, 48 + half * 8);
      sc[ct] = wmma_bf16(qa0, kf0, zero8());
      sc[ct] = wmma_bf16(qa1, kf1, sc[ct]);
    }
    // additive mask (fp32)
#pragma unroll
    for (int ct = 0; ct < 2; ++ct)
#pragma unroll
      for (int r = 0; r < 8; ++r)
        sc[ct][r] += mbase[(size_t)(q0 + r + half * 8) * SEQ + kb + ct * 16 + l16];

    // online softmax update (rows live across 16 lanes of each half-wave)
#pragma unroll
    for (int r = 0; r < 8; ++r) {
      float mx = fmaxf(sc[0][r], sc[1][r]);
#pragma unroll
      for (int off = 8; off >= 1; off >>= 1)
        mx = fmaxf(mx, __shfl_xor(mx, off, 32));
      const float mnew  = fmaxf(mrow[r], mx);
      const float alpha = __expf(mrow[r] - mnew);
      const float p0 = __expf(sc[0][r] - mnew);
      const float p1 = __expf(sc[1][r] - mnew);
      float rs = p0 + p1;
#pragma unroll
      for (int off = 8; off >= 1; off >>= 1)
        rs += __shfl_xor(rs, off, 32);
      lrow[r] = lrow[r] * alpha + rs;
      mrow[r] = mnew;
#pragma unroll
      for (int dt = 0; dt < 4; ++dt) acc[dt][r] *= alpha;
      sc[0][r] = p0; sc[1][r] = p1;
    }

    // C-layout P -> LDS (bf16, 16x32 row-major, per-wave slice)
    bf16_t* pw = &pLds[wave][0];
#pragma unroll
    for (int r = 0; r < 8; ++r) {
      const int row = r + half * 8;
      pw[row * 32 + l16]      = (bf16_t)sc[0][r];
      pw[row * 32 + 16 + l16] = (bf16_t)sc[1][r];
    }
    asm volatile("s_wait_dscnt 0" ::: "memory");  // intra-wave LDS WAR/RAW order
    // re-read P in A-fragment layout
    v16bf pf = load_frag(pw + l16 * 32, half * 8, 16 + half * 8);

    // O += P * V   (V^T rows are K-contiguous)
#pragma unroll
    for (int dt = 0; dt < 4; ++dt) {
      v16bf vf = load_frag(vbase + (size_t)(dt * 16 + l16) * SEQ + kb,
                           half * 8, 16 + half * 8);
      acc[dt] = wmma_bf16(pf, vf, acc[dt]);
    }
    asm volatile("s_wait_dscnt 0" ::: "memory");
  }

  // normalize + store [B,N,H*D] as bf16 for the proj GEMM
#pragma unroll
  for (int r = 0; r < 8; ++r) {
    const int tok = q0 + r + half * 8;
    const float inv = 1.0f / lrow[r];
#pragma unroll
    for (int dt = 0; dt < 4; ++dt) {
      const int ch = h * HDIM + dt * 16 + l16;
      out[((size_t)b * SEQ + tok) * DIM + ch] = (bf16_t)(acc[dt][r] * inv);
    }
  }
}

// ---------------- Proj GEMM: [8192,1024] x [1024,1024]^T + bias -> fp32 -----
__global__ void __launch_bounds__(128) proj_gemm(
    const bf16_t* __restrict__ ab,   // [8192,1024]
    const bf16_t* __restrict__ wb,   // [1024,1024]
    const float*  __restrict__ bias, // [1024]
    float* __restrict__ outp) {      // [8192,1024]
  const int lane = threadIdx.x & 31;
  const int wave = threadIdx.x >> 5;
  const int half = lane >> 4;
  const int l16  = lane & 15;
  const int m0 = blockIdx.x * 128 + wave * 32;
  const int n0 = blockIdx.y * 64;

  v8f acc[2][4];
#pragma unroll
  for (int i = 0; i < 2; ++i)
#pragma unroll
    for (int j = 0; j < 4; ++j) acc[i][j] = zero8();

  for (int k0 = 0; k0 < DIM; k0 += 32) {
    const int kLo = k0 + half * 8, kHi = k0 + 16 + half * 8;
    v16bf a0 = load_frag(ab + (size_t)(m0 + l16) * DIM,      kLo, kHi);
    v16bf a1 = load_frag(ab + (size_t)(m0 + 16 + l16) * DIM, kLo, kHi);
#pragma unroll
    for (int ct = 0; ct < 4; ++ct) {
      v16bf bf = load_frag(wb + (size_t)(n0 + ct * 16 + l16) * DIM, kLo, kHi);
      acc[0][ct] = wmma_bf16(a0, bf, acc[0][ct]);
      acc[1][ct] = wmma_bf16(a1, bf, acc[1][ct]);
    }
  }

#pragma unroll
  for (int rt = 0; rt < 2; ++rt)
#pragma unroll
    for (int ct = 0; ct < 4; ++ct) {
      const float bb = bias[n0 + ct * 16 + l16];
#pragma unroll
      for (int r = 0; r < 8; ++r) {
        const int m = m0 + rt * 16 + r + half * 8;
        outp[(size_t)m * DIM + n0 + ct * 16 + l16] = acc[rt][ct][r] + bb;
      }
    }
}

extern "C" void kernel_launch(void* const* d_in, const int* in_sizes, int n_in,
                              void* d_out, int out_size, void* d_ws, size_t ws_size,
                              hipStream_t stream) {
  const float* x      = (const float*)d_in[0];   // [8,1024,1024]
  const float* mask   = (const float*)d_in[1];   // [8,1,1024,1024]
  const float* w_qkv  = (const float*)d_in[2];   // [3072,1024]
  const float* w_proj = (const float*)d_in[3];   // [1024,1024]
  const float* b_proj = (const float*)d_in[4];   // [1024]

  char* ws = (char*)d_ws;
  size_t off = 0;
  auto alloc = [&](size_t bytes) -> void* {
    void* p = ws + off;
    off = (off + bytes + 255) & ~(size_t)255;
    return p;
  };
  const size_t TOK = (size_t)BATCH * SEQ;            // 8192
  bf16_t* xb   = (bf16_t*)alloc(TOK * DIM * 2);      // 16 MB
  bf16_t* wqb  = (bf16_t*)alloc((size_t)3 * DIM * DIM * 2);
  bf16_t* wpb  = (bf16_t*)alloc((size_t)DIM * DIM * 2);
  bf16_t* qb   = (bf16_t*)alloc(TOK * DIM * 2);
  bf16_t* kbf  = (bf16_t*)alloc(TOK * DIM * 2);
  bf16_t* vtb  = (bf16_t*)alloc(TOK * DIM * 2);
  bf16_t* attn = (bf16_t*)alloc(TOK * DIM * 2);

  const int nx = (int)(TOK * DIM);          // 8388608
  const int nq = 3 * DIM * DIM;             // 3145728
  const int np = DIM * DIM;                 // 1048576
  cvt_f32_bf16<<<(nx + 255) / 256, 256, 0, stream>>>(x, xb, nx);
  cvt_f32_bf16<<<(nq + 255) / 256, 256, 0, stream>>>(w_qkv, wqb, nq);
  cvt_f32_bf16<<<(np + 255) / 256, 256, 0, stream>>>(w_proj, wpb, np);

  qkv_gemm<<<dim3(64, 48), 128, 0, stream>>>(xb, wqb, qb, kbf, vtb);
  flash_attn<<<BATCH * NHEAD * (SEQ / 128), 256, 0, stream>>>(qb, kbf, vtb, mask, attn);
  proj_gemm<<<dim3(64, 16), 128, 0, stream>>>(attn, wpb, b_proj, (float*)d_out);
}
